// SRNNTextEncoder_12902081757505
// MI455X (gfx1250) — matrix operand order repaired and checked by reading, not compiled, not run on hardware
//
#include <hip/hip_runtime.h>
#include <hip/hip_bf16.h>
#include <cstdint>
#include <cstddef>

typedef __bf16 bf16_t;
typedef bf16_t v16bf __attribute__((ext_vector_type(16)));
typedef float  v8f   __attribute__((ext_vector_type(8)));

#define BM 128
#define BN 128
#define BK 32

// ---------------- small helper kernels ----------------

__global__ void cvt_bf16_kernel(const float* __restrict__ s, bf16_t* __restrict__ d, int n) {
  int i = blockIdx.x * blockDim.x + threadIdx.x;
  if (i < n) d[i] = (bf16_t)s[i];
}

__global__ void zero_kernel(uint32_t* __restrict__ p, int n) {
  int i = blockIdx.x * blockDim.x + threadIdx.x;
  if (i < n) p[i] = 0u;
}

__global__ __launch_bounds__(128) void embed_kernel(const int* __restrict__ x,
                                                    const float* __restrict__ emb,
                                                    bf16_t* __restrict__ out) {
  int r = blockIdx.x;                  // 0..255 == b*32 + t
  int tok = x[r];
  const float* src = emb + (size_t)tok * 512;
  bf16_t* dst = out + (size_t)r * 512;
  for (int j = threadIdx.x; j < 512; j += 128) dst[j] = (bf16_t)src[j];
}

// ---------------- bf16 WMMA GEMM:  C[M,N] = A[M,K] * Bw[N,K]^T (+ bias[N]) ----------------

__device__ __forceinline__ v16bf frag_a(const bf16_t* tile, int m_local, int lane) {
  // 16-bit A layout: lanes 0-15 hold K=[h*8,h*8+8)+[16+h*8,..), h = lane>>4
  union { uint4 q[2]; v16bf v; } u;
  const bf16_t* p = tile + m_local * BK;
  int h8 = (lane >> 4) * 8;
  u.q[0] = *(const uint4*)(p + h8);
  u.q[1] = *(const uint4*)(p + 16 + h8);
  return u.v;
}
__device__ __forceinline__ v16bf frag_b(const bf16_t* tile, int n_local, int lane) {
  // 16-bit B layout: lane holds K=[h*8,h*8+8)+[16+h*8,..) of its column, h = lane>>4
  union { uint4 q[2]; v16bf v; } u;
  const bf16_t* p = tile + n_local * BK + ((lane >> 4) << 3);
  u.q[0] = *(const uint4*)(p);
  u.q[1] = *(const uint4*)(p + 16);
  return u.v;
}

__global__ __launch_bounds__(256) void gemm_bf16_kernel(
    const bf16_t* __restrict__ A, const bf16_t* __restrict__ Bw,
    const float* __restrict__ bias, float* __restrict__ C,
    int M, int N, int K) {
  __shared__ alignas(16) bf16_t As[2][BM * BK];
  __shared__ alignas(16) bf16_t Bs[2][BN * BK];

  const int tid  = threadIdx.x;
  const int lane = tid & 31;
  const int wave = tid >> 5;            // 8 waves
  const int wm   = (wave & 3) * 32;     // wave tile: 32 rows x 64 cols
  const int wn   = (wave >> 2) * 64;
  const long bm  = (long)blockIdx.y * BM;
  const long bn  = (long)blockIdx.x * BN;

  const int r0 = tid >> 2;              // 0..63: staging row
  const int q0 = (tid & 3) * 8;         // 16-byte chunk within 32-elem row

  v8f acc[2][4];
#pragma unroll
  for (int i = 0; i < 2; ++i)
#pragma unroll
    for (int j = 0; j < 4; ++j)
#pragma unroll
      for (int r = 0; r < 8; ++r) acc[i][j][r] = 0.0f;

  uint4 ga0, ga1, gb0, gb1;
  auto gload = [&](int kt) {
    const bf16_t* a = A + (bm + r0) * (long)K + kt * BK + q0;
    ga0 = *(const uint4*)a;
    ga1 = *(const uint4*)(a + 64L * K);
    const bf16_t* b = Bw + (bn + r0) * (long)K + kt * BK + q0;
    gb0 = *(const uint4*)b;
    gb1 = *(const uint4*)(b + 64L * K);
  };
  auto sstore = [&](int buf) {
    *(uint4*)&As[buf][r0 * BK + q0]        = ga0;
    *(uint4*)&As[buf][(r0 + 64) * BK + q0] = ga1;
    *(uint4*)&Bs[buf][r0 * BK + q0]        = gb0;
    *(uint4*)&Bs[buf][(r0 + 64) * BK + q0] = gb1;
  };

  const int KT = K / BK;
  gload(0);
  sstore(0);
  for (int kt = 0; kt < KT; ++kt) {
    __syncthreads();
    const int buf = kt & 1;
    const bool pf = (kt + 1) < KT;
    if (pf) gload(kt + 1);               // global loads in flight while WMMAs run

    v16bf af[2], bv[4];
#pragma unroll
    for (int mt = 0; mt < 2; ++mt) af[mt] = frag_a(&As[buf][0], wm + mt * 16 + (lane & 15), lane);
#pragma unroll
    for (int nt = 0; nt < 4; ++nt) bv[nt] = frag_b(&Bs[buf][0], wn + nt * 16 + (lane & 15), lane);
#pragma unroll
    for (int mt = 0; mt < 2; ++mt)
#pragma unroll
      for (int nt = 0; nt < 4; ++nt)
        acc[mt][nt] = __builtin_amdgcn_wmma_f32_16x16x32_bf16(
            false, af[mt], false, bv[nt], (short)0, acc[mt][nt], false, false);

    if (pf) sstore(buf ^ 1);
  }

  // epilogue: f32 C/D layout: n = lane&15, m = (lane>>4)*8 + r
#pragma unroll
  for (int mt = 0; mt < 2; ++mt)
#pragma unroll
    for (int nt = 0; nt < 4; ++nt) {
      const long col = bn + wn + nt * 16 + (lane & 15);
      const float bvs = bias ? bias[col] : 0.0f;
#pragma unroll
      for (int r = 0; r < 8; ++r) {
        const long row = bm + wm + mt * 16 + (lane >> 4) * 8 + r;
        C[row * (long)N + col] = acc[mt][nt][r] + bvs;
      }
    }
}

// ---------------- LSTM cell + running-sum + span-mean scatter ----------------

__device__ __forceinline__ float sigm(float x) { return 1.0f / (1.0f + __expf(-x)); }

__global__ __launch_bounds__(128) void lstm_cell_kernel(
    const float* __restrict__ gates_f, const float* __restrict__ gates_b,
    const float* __restrict__ xproj_f, const float* __restrict__ xproj_b,
    const float* __restrict__ bih_f, const float* __restrict__ bhh_f,
    const float* __restrict__ bih_b, const float* __restrict__ bhh_b,
    float* __restrict__ c_f, float* __restrict__ c_b,
    bf16_t* __restrict__ h_f, bf16_t* __restrict__ h_b,
    float* __restrict__ rs_f, float* __restrict__ rs_b,
    bf16_t* __restrict__ meanvec, int tau) {
  const int dir = blockIdx.y;                 // 0 fwd, 1 bwd
  const int seq = blockIdx.x;                 // b*32 + idx (idx = s fwd, e bwd)
  const int b = seq >> 5, idx = seq & 31;
  if (dir == 0) { if (tau < idx) return; } else { if (tau > idx) return; }
  const int t = (dir == 0) ? tau : (idx - tau);
  const int s = (dir == 0) ? idx : t;
  const int e = (dir == 0) ? tau : idx;

  const float* gr = (dir ? gates_b : gates_f) + (size_t)seq * 2048;
  const float* xp = (dir ? xproj_b : xproj_f) + (size_t)(b * 32 + t) * 2048;
  const float* bi = dir ? bih_b : bih_f;
  const float* bh = dir ? bhh_b : bhh_f;
  float*  cc = (dir ? c_b : c_f)   + (size_t)seq * 512;
  bf16_t* hh = (dir ? h_b : h_f)   + (size_t)seq * 512;
  float*  rs = (dir ? rs_b : rs_f) + (size_t)seq * 512;

  long outrow = -1; float invL = 0.0f;
  if (e > s) {                                // spans of length >= 2 only
    int k = e - s;
    int chunkoff = (k - 1) * 32 - ((k - 1) * k) / 2;
    outrow = (long)b * 496 + chunkoff + s;
    invL = 1.0f / (float)(k + 1);
  }

#pragma unroll
  for (int q = 0; q < 4; ++q) {
    int j = threadIdx.x + q * 128;
    float gi = gr[j]        + xp[j]        + bi[j]        + bh[j];
    float gf = gr[512 + j]  + xp[512 + j]  + bi[512 + j]  + bh[512 + j];
    float gg = gr[1024 + j] + xp[1024 + j] + bi[1024 + j] + bh[1024 + j];
    float go = gr[1536 + j] + xp[1536 + j] + bi[1536 + j] + bh[1536 + j];
    float c = sigm(gf) * cc[j] + sigm(gi) * tanhf(gg);
    float h = sigm(go) * tanhf(c);
    cc[j] = c;
    hh[j] = (bf16_t)h;
    float r = rs[j] + h;
    rs[j] = r;
    if (outrow >= 0)
      meanvec[outrow * 1024 + dir * 512 + j] = (bf16_t)(r * invL);
  }
}

// ---------------- host orchestration ----------------

extern "C" void kernel_launch(void* const* d_in, const int* in_sizes, int n_in,
                              void* d_out, int out_size, void* d_ws, size_t ws_size,
                              hipStream_t stream) {
  (void)in_sizes; (void)n_in; (void)out_size; (void)ws_size;
  const int*   x     = (const int*)d_in[0];
  const float* emb   = (const float*)d_in[2];
  const float* Wih_f = (const float*)d_in[3];
  const float* Whh_f = (const float*)d_in[4];
  const float* bih_f = (const float*)d_in[5];
  const float* bhh_f = (const float*)d_in[6];
  const float* Wih_b = (const float*)d_in[7];
  const float* Whh_b = (const float*)d_in[8];
  const float* bih_b = (const float*)d_in[9];
  const float* bhh_b = (const float*)d_in[10];
  const float* Wout  = (const float*)d_in[11];
  const float* bout  = (const float*)d_in[12];

  const int G = 2048, W = 512, H = 512, K2 = 1024;
  const int NW  = G * W;          // 2048*512 per ih/hh weight
  const int NWO = 7680 * K2;      // Wout
  const int SEQ = 256;            // B*N
  const int ROWS = 3968;          // B * 496 span rows

  char* w = (char*)d_ws;
  size_t off = 0;
  auto alloc = [&](size_t bytes) { size_t o = off; off = (off + bytes + 255) & ~(size_t)255; return o; };
  bf16_t* wihf = (bf16_t*)(w + alloc((size_t)NW * 2));
  bf16_t* whhf = (bf16_t*)(w + alloc((size_t)NW * 2));
  bf16_t* wihb = (bf16_t*)(w + alloc((size_t)NW * 2));
  bf16_t* whhb = (bf16_t*)(w + alloc((size_t)NW * 2));
  bf16_t* wout = (bf16_t*)(w + alloc((size_t)NWO * 2));
  bf16_t* embb = (bf16_t*)(w + alloc((size_t)SEQ * W * 2));
  float*  xpf  = (float*)(w + alloc((size_t)SEQ * G * 4));
  float*  xpb  = (float*)(w + alloc((size_t)SEQ * G * 4));
  float*  gtf  = (float*)(w + alloc((size_t)SEQ * G * 4));
  float*  gtb  = (float*)(w + alloc((size_t)SEQ * G * 4));
  bf16_t* mean = (bf16_t*)(w + alloc((size_t)ROWS * K2 * 2));
  // contiguous zero-init state block: c_f,c_b,rs_f,rs_b (f32) then h_f,h_b (bf16)
  size_t state_off = off;
  float*  c_f  = (float*)(w + alloc((size_t)SEQ * H * 4));
  float*  c_b  = (float*)(w + alloc((size_t)SEQ * H * 4));
  float*  rs_f = (float*)(w + alloc((size_t)SEQ * H * 4));
  float*  rs_b = (float*)(w + alloc((size_t)SEQ * H * 4));
  bf16_t* h_f  = (bf16_t*)(w + alloc((size_t)SEQ * H * 2));
  bf16_t* h_b  = (bf16_t*)(w + alloc((size_t)SEQ * H * 2));
  size_t state_bytes = off - state_off;

  auto cvt = [&](const float* s, bf16_t* d, int n) {
    cvt_bf16_kernel<<<(n + 255) / 256, 256, 0, stream>>>(s, d, n);
  };
  cvt(Wih_f, wihf, NW);
  cvt(Whh_f, whhf, NW);
  cvt(Wih_b, wihb, NW);
  cvt(Whh_b, whhb, NW);
  cvt(Wout,  wout, NWO);

  embed_kernel<<<SEQ, 128, 0, stream>>>(x, emb, embb);

  int zwords = (int)(state_bytes / 4);
  zero_kernel<<<(zwords + 255) / 256, 256, 0, stream>>>((uint32_t*)(w + state_off), zwords);

  auto gemm = [&](const bf16_t* A, const bf16_t* Bw, const float* bias, float* C,
                  int M, int N, int K) {
    dim3 grid(N / BN, M / BM);
    gemm_bf16_kernel<<<grid, 256, 0, stream>>>(A, Bw, bias, C, M, N, K);
  };

  // token projections x @ Wih^T (bias folded into cell kernel)
  gemm(embb, wihf, nullptr, xpf, SEQ, G, W);
  gemm(embb, wihb, nullptr, xpb, SEQ, G, W);

  // 32 sequential biLSTM steps: h @ Whh^T per direction, then fused cell update
  for (int tau = 0; tau < 32; ++tau) {
    gemm(h_f, whhf, nullptr, gtf, SEQ, G, H);
    gemm(h_b, whhb, nullptr, gtb, SEQ, G, H);
    lstm_cell_kernel<<<dim3(SEQ, 2), 128, 0, stream>>>(
        gtf, gtb, xpf, xpb, bih_f, bhh_f, bih_b, bhh_b,
        c_f, c_b, h_f, h_b, rs_f, rs_b, mean, tau);
  }

  // final projection of span means: [3968,1024] x [1024,7680] + bout -> d_out
  gemm(mean, wout, bout, (float*)d_out, ROWS, 7680, K2);
}
